// GraphEmbedding_83245056131906
// MI455X (gfx1250) — matrix operand order repaired and checked by reading, not compiled, not run on hardware
//
#include <hip/hip_runtime.h>
#include <math.h>

typedef __bf16 bf16_t;
typedef __attribute__((ext_vector_type(16))) __bf16 v16bf;
typedef __attribute__((ext_vector_type(8)))  float  v8f;

#define NTHREADS 512
#define NWAVES   16
#define QB       8      // queries per block
#define NN       20     // neighbors per query
#define DD       172    // node feature dim (= DE)
#define EE       272    // attn embed dim E
#define EP       288    // E padded to mult of 32
#define DKQ      444    // kv input dim
#define DKP      448    // kv dim padded to mult of 32
#define HD       136    // head dim (E/2)
#define KVROWS   160    // QB*NN (10 exact 16-row tiles; 5 exact 32-k chunks)
#define SLD      160    // scores leading dim
#define ALD      160    // attention-weight leading dim (k for U GEMM, no pad)
#define H1LD     192    // W2 input dim padded (172 -> 192)
#define D1P      176    // W1/W2 output rows padded (172 -> 176)

// async gather pipeline
#define ROWT     (2 * KVROWS)   // 320 row tasks: 160 neighbor rows + 160 edge rows
#define CH       16             // rows per chunk
#define NCH      (ROWT / CH)    // 20 chunks
#define OPS      (CH * 43)      // 688 b128 transfers per chunk (172 f32 = 43 x 16B)
#define PT       2              // async ops per thread per chunk (512*2 >= 688)
#define STGB     (CH * DD * 4)  // 11008 B staging per buffer (row stride 688B, 16B aligned)

#define VZ {0.f,0.f,0.f,0.f,0.f,0.f,0.f,0.f}

// ---------------- dynamic LDS layout (phase-aliased) ----------------
constexpr int SM_KV   = 0;
constexpr int SM_RA   = SM_KV   + KVROWS * DKP * 2;     // 143360
constexpr int SM_RB   = SM_RA   + 16 * DKP * 2;         // 157696
constexpr int SM_RC   = SM_RB   + 16 * EP * 2;          // 166912
constexpr int SM_S    = SM_RC   + 16 * EP * 2;          // 176128
constexpr int SM_SRCF = SM_S    + 16 * SLD * 4;         // 186368
constexpr int SM_NID  = SM_SRCF + QB * DD * 4;          // 191872
constexpr int SM_EID  = SM_NID  + KVROWS * 4;
constexpr int SM_DT   = SM_EID  + KVROWS * 4;
constexpr int SM_MSK  = SM_DT   + KVROWS * 4;
constexpr int SM_SC   = SM_MSK  + KVROWS * 4;
constexpr int SM_TS   = SM_SC   + 16 * 4;
constexpr int SM_ALLM = SM_TS   + QB * 4;
constexpr int SM_SID  = SM_ALLM + QB * 4;
constexpr int SM_JUNK = SM_SID  + QB * 4;               // 16B dummy async dst
constexpr int SM_TOTAL= SM_JUNK + 16;                   // ~190 KB
static_assert(2 * STGB <= 16 * DKP * 2 + 16 * EP * 2, "staging must fit in RA+RB");
static_assert(SM_JUNK % 16 == 0, "junk slot must be 16B aligned");

// ---------------- WMMA operand loaders ----------------
__device__ inline void ldA(v16bf& a, const bf16_t* __restrict__ ap) {
#pragma unroll
  for (int i = 0; i < 16; ++i) a[i] = ap[i];
}
__device__ inline void ldB(v16bf& b, const bf16_t* __restrict__ bp) {
#pragma unroll
  for (int i = 0; i < 16; ++i) b[i] = bp[i];
}
__device__ inline void ldBs(v16bf& b, const bf16_t* __restrict__ bp, int ldb) {
#pragma unroll
  for (int i = 0; i < 16; ++i) b[i] = bp[(size_t)i * ldb];
}

// Software-pipelined k-loop: loads for chunk k+1 fly while WMMA for chunk k executes.
// Tail loads clamp to chunk 0 (harmless re-read) so loads stay unconditional;
// only the second WMMA is under a wave-uniform guard (EXEC stays all-ones).
__device__ inline v8f wmma_acc_rowB(const bf16_t* __restrict__ A, int lda,
                                    const bf16_t* __restrict__ B, int ldb,
                                    int kTot, int nbase, int lane, v8f acc) {
  const int m = lane & 15, g = lane >> 4;
  const bf16_t* ap = A + m * lda + g * 16;
  const bf16_t* bp = B + (size_t)(nbase + m) * ldb + g * 16;
  v16bf a0, b0, a1, b1;
  ldA(a0, ap); ldB(b0, bp);
  for (int kb = 0; kb < kTot; kb += 64) {
    int k1 = (kb + 32 < kTot) ? kb + 32 : 0;
    ldA(a1, ap + k1); ldB(b1, bp + k1);
    acc = __builtin_amdgcn_wmma_f32_16x16x32_bf16(false, a0, false, b0,
                                                  (short)0, acc, false, false);
    int k2 = (kb + 64 < kTot) ? kb + 64 : 0;
    ldA(a0, ap + k2); ldB(b0, bp + k2);
    if (kb + 32 < kTot)
      acc = __builtin_amdgcn_wmma_f32_16x16x32_bf16(false, a1, false, b1,
                                                    (short)0, acc, false, false);
  }
  return acc;
}

// B stored k-major: Bk[k*ldb + n] (strided lane reads; used for U = A_att @ kv).
__device__ inline v8f wmma_acc_strideB(const bf16_t* __restrict__ A, int lda,
                                       const bf16_t* __restrict__ Bk, int ldb,
                                       int kTot, int nbase, int lane, v8f acc) {
  const int m = lane & 15, g = lane >> 4;
  const bf16_t* ap = A + m * lda + g * 16;
  const bf16_t* bp = Bk + (size_t)(g * 16) * ldb + nbase + m;
  v16bf a0, b0, a1, b1;
  ldA(a0, ap); ldBs(b0, bp, ldb);
  for (int kb = 0; kb < kTot; kb += 64) {
    int k1 = (kb + 32 < kTot) ? kb + 32 : 0;
    ldA(a1, ap + k1); ldBs(b1, bp + (size_t)k1 * ldb, ldb);
    acc = __builtin_amdgcn_wmma_f32_16x16x32_bf16(false, a0, false, b0,
                                                  (short)0, acc, false, false);
    int k2 = (kb + 64 < kTot) ? kb + 64 : 0;
    ldA(a0, ap + k2); ldBs(b0, bp + (size_t)k2 * ldb, ldb);
    if (kb + 32 < kTot)
      acc = __builtin_amdgcn_wmma_f32_16x16x32_bf16(false, a1, false, b1,
                                                    (short)0, acc, false, false);
  }
  return acc;
}

// ---------------- weight packing ----------------
__global__ void pack_bf16(const float* __restrict__ s, bf16_t* __restrict__ d,
                          int rows, int rowsPad, int cols, int ldd) {
  int i = blockIdx.x * blockDim.x + threadIdx.x;
  if (i >= rowsPad * ldd) return;
  int r = i / ldd, c = i % ldd;
  float v = (r < rows && c < cols) ? s[(size_t)r * cols + c] : 0.f;
  d[i] = (bf16_t)v;
}

__global__ void pack_bf16_T(const float* __restrict__ s, bf16_t* __restrict__ d,
                            int srows, int scols, int drows, int ldd) {
  int i = blockIdx.x * blockDim.x + threadIdx.x;
  if (i >= drows * ldd) return;
  int c = i / ldd, r = i % ldd;              // d[c][r] = s[r][c]
  float v = (c < scols && r < srows) ? s[(size_t)r * scols + c] : 0.f;
  d[i] = (bf16_t)v;
}

// ---------------- async row-gather (memory/edge rows -> LDS staging) ----------------
__device__ inline const float* task_src(int r, int gatherNeigh, int q0,
                                        const float* __restrict__ memory,
                                        const float* __restrict__ nsrc,
                                        const float* __restrict__ edgef,
                                        const int* __restrict__ NID,
                                        const int* __restrict__ EID) {
  if (r < KVROWS) {
    if (gatherNeigh) return memory + (size_t)NID[r] * DD;
    return nsrc + ((size_t)(q0 + r / NN) * NN + (r % NN)) * DD;
  }
  return edgef + (size_t)EID[r - KVROWS] * DD;
}

__device__ inline void issue_chunk(int c, int buf, int tid, int gatherNeigh, int q0,
                                   const float* __restrict__ memory,
                                   const float* __restrict__ nsrc,
                                   const float* __restrict__ edgef,
                                   const int* __restrict__ NID,
                                   const int* __restrict__ EID) {
  const int base = c * CH;
  const unsigned stg0 = (unsigned)(SM_RA + buf * STGB);
#pragma unroll
  for (int t = 0; t < PT; ++t) {
    int o = tid + t * NTHREADS;
    unsigned lds = SM_JUNK;
    const float* src = memory;                 // dummy (safe) transfer
    if (o < OPS) {
      int row = o / 43, j = o % 43;            // j-th 16B of the 688B row
      src = task_src(base + row, gatherNeigh, q0, memory, nsrc, edgef, NID, EID) + j * 4;
      lds = stg0 + (unsigned)(row * (DD * 4) + j * 16);
    }
    unsigned long long ga = (unsigned long long)(const void*)src;
    // CDNA5 async DMA: LDS[vdst] = MEM[vaddr], tracked by ASYNCcnt
    asm volatile("global_load_async_to_lds_b128 %0, %1, off"
                 :: "v"(lds), "v"(ga) : "memory");
  }
}

// ---------------- fused temporal-attention aggregate ----------------
__global__ __launch_bounds__(NTHREADS)
void tgn_agg(int numQ, int tsDiv, int gatherNeigh,
             const float* __restrict__ memory, const float* __restrict__ edgef,
             const float* __restrict__ nsrc,   // memory (L1) or emb1 (L2)
             const int* __restrict__ src_ids, const float* __restrict__ tstamps,
             const int* __restrict__ nbrs, const int* __restrict__ eidx,
             const float* __restrict__ etimes,
             const float* __restrict__ w_time, const float* __restrict__ b_time,
             const bf16_t* __restrict__ wq, const float* __restrict__ bq,
             const bf16_t* __restrict__ wkT, const float* __restrict__ bk,
             const bf16_t* __restrict__ wv, const float* __restrict__ bv,
             const bf16_t* __restrict__ wo, const float* __restrict__ bo,
             const bf16_t* __restrict__ w1, const float* __restrict__ b1,
             const bf16_t* __restrict__ w2, const float* __restrict__ b2,
             float* __restrict__ out) {
  extern __shared__ unsigned char smem[];
  bf16_t* KV  = (bf16_t*)(smem + SM_KV);
  bf16_t* RA  = (bf16_t*)(smem + SM_RA);
  bf16_t* RB  = (bf16_t*)(smem + SM_RB);
  bf16_t* RC  = (bf16_t*)(smem + SM_RC);
  float*  SS  = (float*)(smem + SM_S);
  float*  SRCF= (float*)(smem + SM_SRCF);
  int*    NID = (int*)(smem + SM_NID);
  int*    EID = (int*)(smem + SM_EID);
  float*  DTm = (float*)(smem + SM_DT);
  int*    MSK = (int*)(smem + SM_MSK);
  float*  SC  = (float*)(smem + SM_SC);
  int*    ALLM= (int*)(smem + SM_ALLM);
  int*    SID = (int*)(smem + SM_SID);

  const int tid = threadIdx.x, lane = tid & 31, wav = tid >> 5;
  const int m = lane & 15, g = lane >> 4;
  const int q0 = blockIdx.x * QB;

  if (wav == 0) {  // warm weight stream into L2 (gfx1250 global_prefetch path)
    __builtin_prefetch(wq + (size_t)lane * 2048, 0, 1);
    __builtin_prefetch(wv + (size_t)lane * 2048, 0, 1);
  }

  // ---- Phase A0: metadata ----
  if (tid < KVROWS) {
    int r = tid, qi = r / NN, n = r % NN, gq = q0 + qi;
    int nid = 0, eid = 0; float et = 0.f, ts = 0.f;
    if (gq < numQ) {
      nid = nbrs[(size_t)gq * NN + n];
      eid = eidx[(size_t)gq * NN + n];
      et  = etimes[(size_t)gq * NN + n];
      ts  = tstamps[gq / tsDiv];
    }
    NID[r] = nid; EID[r] = eid; DTm[r] = ts - et;
    MSK[r] = (gq < numQ) ? (nid == 0) : 1;
  }
  if (tid < QB) {
    int gq = q0 + tid;
    SID[tid] = (gq < numQ) ? src_ids[gq] : 0;
  }
  __syncthreads();

  // ---- Phase A1: async row-gather pipeline (neighbor + edge rows -> KV bf16) ----
  issue_chunk(0, 0, tid, gatherNeigh, q0, memory, nsrc, edgef, NID, EID);

  // Overlap with first DMA: time encodings + pads + masks + source features.
  if (tid < QB) {
    int am = 1;
    for (int n = 0; n < NN; ++n) am &= MSK[tid * NN + n];
    ALLM[tid] = am;
  }
  for (int i = tid; i < KVROWS * (EE - DD); i += NTHREADS) {
    int r = i / (EE - DD), j = i % (EE - DD);
    KV[r * DKP + DD + j] = (bf16_t)cosf(DTm[r] * w_time[j] + b_time[j]);
  }
  for (int i = tid; i < KVROWS * (DKP - DKQ); i += NTHREADS)
    KV[(i / (DKP - DKQ)) * DKP + DKQ + (i % (DKP - DKQ))] = (bf16_t)0.f;
  for (int i = tid; i < QB * DD; i += NTHREADS) {
    int qi = i / DD, c = i % DD;
    SRCF[i] = memory[(size_t)SID[qi] * DD + c];
  }

  for (int c = 0; c < NCH; ++c) {
    if (c + 1 < NCH) {
      issue_chunk(c + 1, (c + 1) & 1, tid, gatherNeigh, q0, memory, nsrc, edgef, NID, EID);
      asm volatile("s_wait_asynccnt 0x2" ::: "memory");  // retire chunk c, keep c+1 flying
    } else {
      asm volatile("s_wait_asynccnt 0x0" ::: "memory");
    }
    __syncthreads();
    const float* stg = (const float*)(smem + SM_RA + (size_t)(c & 1) * STGB);
    for (int i = tid; i < CH * DD; i += NTHREADS) {
      int row = i / DD, cc = i % DD;
      int r = c * CH + row;
      int kvrow = (r < KVROWS) ? r : (r - KVROWS);
      int colb  = (r < KVROWS) ? 0 : EE;
      KV[kvrow * DKP + colb + cc] = (bf16_t)stg[row * DD + cc];
    }
    __syncthreads();
  }

  // ---- Phase A2: q_in -> RA (ld DKP); zero RB pad cols [272,288) ----
  for (int i = tid; i < 16 * EP; i += NTHREADS) {
    int q = i / EP, c = i % EP;
    float v = 0.f;
    if (q < QB) {
      if (c < DD) v = SRCF[q * DD + c];
      else if (c < EE) v = cosf(b_time[c - DD]);
    }
    RA[q * DKP + c] = (bf16_t)v;
  }
  for (int i = tid; i < 16 * (EP - EE); i += NTHREADS) {
    int q = i / (EP - EE), c = EE + i % (EP - EE);
    RB[q * EP + c] = (bf16_t)0.f;
  }
  __syncthreads();

  // ---- G1: Q = q_in @ Wq^T + bq, head-split into T_in (RB, all 16 rows live) ----
  {
    v8f acc[2] = {VZ, VZ};
    for (int t = 0; t < 2; ++t) {
      int tile = wav + t * NWAVES;
      if (tile < 17) acc[t] = wmma_acc_rowB(RA, DKP, wq, EP, EP, tile * 16, lane, acc[t]);
    }
    __syncthreads();
    for (int t = 0; t < 2; ++t) {
      int tile = wav + t * NWAVES;
      if (tile < 17) {
        int e = tile * 16 + m;
#pragma unroll
        for (int v = 0; v < 8; ++v) {
          int q = v + 8 * g;
          if (q < 8) {
            float val = acc[t][v] + bq[e];
            int h = (e >= HD) ? 1 : 0;
            RB[(h * 8 + q) * EP + e] = (bf16_t)val;
            RB[((1 - h) * 8 + q) * EP + e] = (bf16_t)0.f;
          }
        }
      }
    }
    __syncthreads();
  }

  // ---- folded key-bias SC[r] = dot(T_in[r], bk); G2: T = T_in @ Wk ----
  if (tid < 16) {
    float a = 0.f;
    for (int e = 0; e < EE; ++e) a += (float)RB[tid * EP + e] * bk[e];
    SC[tid] = a;
  }
  {
    v8f acc[2] = {VZ, VZ};
    for (int t = 0; t < 2; ++t) {
      int tile = wav + t * NWAVES;
      if (tile < 28) acc[t] = wmma_acc_rowB(RB, EP, wkT, EP, EP, tile * 16, lane, acc[t]);
    }
    __syncthreads();
    for (int t = 0; t < 2; ++t) {
      int tile = wav + t * NWAVES;
      if (tile < 28) {
        int c = tile * 16 + m;
#pragma unroll
        for (int v = 0; v < 8; ++v) RA[(v + 8 * g) * DKP + c] = (bf16_t)acc[t][v];
      }
    }
    __syncthreads();
  }

  // ---- G3: S = T @ kv^T  ([16][160] f32) ----
  {
    v8f acc0 = VZ;
    if (wav < 10) acc0 = wmma_acc_rowB(RA, DKP, KV, DKP, DKP, wav * 16, lane, acc0);
    __syncthreads();
    if (wav < 10) {
      int c = wav * 16 + m;
#pragma unroll
      for (int v = 0; v < 8; ++v) SS[(v + 8 * g) * SLD + c] = acc0[v];
    }
    __syncthreads();
  }

  // ---- softmax (mask + scale + folded bk) -> A_att bf16 [16][160] in RC ----
  for (int i = tid; i < 16 * ALD; i += NTHREADS) RC[i] = (bf16_t)0.f;
  __syncthreads();
  if (lane == 0) {          // 16 waves = 8 queries x 2 heads
    int q = wav >> 1, h = wav & 1, row = h * 8 + q;
    if (q0 + q < numQ) {
      const float sscale = 0.08574929257125442f;  // 1/sqrt(136)
      float vals[NN], mx = -3.4e38f;
      for (int n = 0; n < NN; ++n) {
        float v = (SS[row * SLD + q * NN + n] + SC[row]) * sscale;
        if (MSK[q * NN + n]) v = -1e9f;
        vals[n] = v; mx = fmaxf(mx, v);
      }
      float s = 0.f;
      for (int n = 0; n < NN; ++n) { float e = __expf(vals[n] - mx); vals[n] = e; s += e; }
      float inv = 1.f / s;
      for (int n = 0; n < NN; ++n) RC[row * ALD + q * NN + n] = (bf16_t)(vals[n] * inv);
    }
  }
  __syncthreads();

  // ---- G4: U = A_att @ kv  ([16][448] -> RA), k = 160 exact ----
  {
    v8f acc[2] = {VZ, VZ};
    for (int t = 0; t < 2; ++t) {
      int tile = wav + t * NWAVES;
      if (tile < 28) acc[t] = wmma_acc_strideB(RC, ALD, KV, DKP, ALD, tile * 16, lane, acc[t]);
    }
    __syncthreads();
    for (int t = 0; t < 2; ++t) {
      int tile = wav + t * NWAVES;
      if (tile < 28) {
        int c = tile * 16 + m;
#pragma unroll
        for (int v = 0; v < 8; ++v) RA[(v + 8 * g) * DKP + c] = (bf16_t)acc[t][v];
      }
    }
    __syncthreads();
  }

  // ---- G5: O' = U @ Wv^T  ([16][272] -> RB) ----
  {
    v8f acc[2] = {VZ, VZ};
    for (int t = 0; t < 2; ++t) {
      int tile = wav + t * NWAVES;
      if (tile < 17) acc[t] = wmma_acc_rowB(RA, DKP, wv, DKP, DKP, tile * 16, lane, acc[t]);
    }
    __syncthreads();
    for (int t = 0; t < 2; ++t) {
      int tile = wav + t * NWAVES;
      if (tile < 17) {
        int e = tile * 16 + m;
#pragma unroll
        for (int v = 0; v < 8; ++v) RB[(v + 8 * g) * EP + e] = (bf16_t)acc[t][v];
      }
    }
    __syncthreads();
  }

  // ---- merge heads + bv (softmax rows sum to 1) -> O merged in RC [16][288] ----
  for (int i = tid; i < 16 * EP; i += NTHREADS) {
    int q = i / EP, e = i % EP;
    float v = 0.f;
    if (q < QB && e < EE) {
      int h = (e >= HD) ? 1 : 0;
      v = (float)RB[(h * 8 + q) * EP + e] + bv[e];
    }
    RC[q * EP + e] = (bf16_t)v;
  }
  __syncthreads();

  // ---- G6: o = O @ Wo^T + bo (zero fully-masked) -> h_in RA = [o || src_feat] ----
  {
    v8f acc[2] = {VZ, VZ};
    for (int t = 0; t < 2; ++t) {
      int tile = wav + t * NWAVES;
      if (tile < 17) acc[t] = wmma_acc_rowB(RC, EP, wo, EP, EP, tile * 16, lane, acc[t]);
    }
    __syncthreads();
    for (int t = 0; t < 2; ++t) {
      int tile = wav + t * NWAVES;
      if (tile < 17) {
        int e = tile * 16 + m;
#pragma unroll
        for (int v = 0; v < 8; ++v) {
          int q = v + 8 * g;
          float val = 0.f;
          if (q < QB && !ALLM[q]) val = acc[t][v] + bo[e];
          RA[q * DKP + e] = (bf16_t)val;
        }
      }
    }
    for (int i = tid; i < 16 * (DKP - EE); i += NTHREADS) {
      int q = i / (DKP - EE), c = EE + i % (DKP - EE);
      float v = 0.f;
      if (q < QB && c < EE + DD) v = SRCF[q * DD + (c - EE)];
      RA[q * DKP + c] = (bf16_t)v;
    }
    __syncthreads();
  }

  // ---- G7: H1 = relu(h_in @ W1^T + b1)  ([16][192] -> RB) ----
  {
    v8f acc0 = VZ;
    if (wav < 11) acc0 = wmma_acc_rowB(RA, DKP, w1, DKP, DKP, wav * 16, lane, acc0);
    __syncthreads();
    if (wav < 11) {
      int c = wav * 16 + m;
#pragma unroll
      for (int v = 0; v < 8; ++v) {
        int q = v + 8 * g;
        float val = 0.f;
        if (q < QB && c < DD) val = fmaxf(acc0[v] + b1[c], 0.f);
        RB[q * H1LD + c] = (bf16_t)val;
      }
    }
    for (int i = tid; i < 16 * (H1LD - D1P); i += NTHREADS)
      RB[(i / (H1LD - D1P)) * H1LD + D1P + (i % (H1LD - D1P))] = (bf16_t)0.f;
    __syncthreads();
  }

  // ---- G8: OUT = H1 @ W2^T + b2 -> global ----
  {
    v8f acc0 = VZ;
    if (wav < 11) {
      acc0 = wmma_acc_rowB(RB, H1LD, w2, H1LD, H1LD, wav * 16, lane, acc0);
      int c = wav * 16 + m;
#pragma unroll
      for (int v = 0; v < 8; ++v) {
        int q = v + 8 * g;
        if (q < QB && c < DD && (q0 + q) < numQ)
          out[(size_t)(q0 + q) * DD + c] = acc0[v] + b2[c];
      }
    }
  }
}

// ---------------- host launcher ----------------
extern "C" void kernel_launch(void* const* d_in, const int* in_sizes, int n_in,
                              void* d_out, int out_size, void* d_ws, size_t ws_size,
                              hipStream_t stream) {
  const float* memory = (const float*)d_in[0];
  const float* edgef  = (const float*)d_in[1];
  const int*   srcN   = (const int*)d_in[2];
  const float* tstmp  = (const float*)d_in[3];
  const int*   nbr1   = (const int*)d_in[4];
  const int*   eidx1  = (const int*)d_in[5];
  const float* etim1  = (const float*)d_in[6];
  const int*   nbr2   = (const int*)d_in[7];
  const int*   eidx2  = (const int*)d_in[8];
  const float* etim2  = (const float*)d_in[9];
  const float* w_time = (const float*)d_in[10];
  const float* b_time = (const float*)d_in[11];
  const float* Wq = (const float*)d_in[12]; const float* bq = (const float*)d_in[13];
  const float* Wk = (const float*)d_in[14]; const float* bk = (const float*)d_in[15];
  const float* Wv = (const float*)d_in[16]; const float* bv = (const float*)d_in[17];
  const float* Wo = (const float*)d_in[18]; const float* bo = (const float*)d_in[19];
  const float* W1 = (const float*)d_in[20]; const float* b1 = (const float*)d_in[21];
  const float* W2 = (const float*)d_in[22]; const float* b2 = (const float*)d_in[23];
  float* out = (float*)d_out;

  char* ws = (char*)d_ws;
  size_t off = 0;
  auto take = [&](size_t bytes) {
    char* p = ws + off;
    off = (off + bytes + 255) & ~(size_t)255;
    return p;
  };
  bf16_t* wq_bf  = (bf16_t*)take((size_t)2 * 272 * 288 * 2);
  bf16_t* wkT_bf = (bf16_t*)take((size_t)2 * 448 * 288 * 2);
  bf16_t* wv_bf  = (bf16_t*)take((size_t)2 * 272 * 448 * 2);
  bf16_t* wo_bf  = (bf16_t*)take((size_t)2 * 272 * 288 * 2);
  bf16_t* w1_bf  = (bf16_t*)take((size_t)2 * 176 * 448 * 2);
  bf16_t* w2_bf  = (bf16_t*)take((size_t)2 * 176 * 192 * 2);
  float*  emb1   = (float*)take((size_t)20000 * 172 * 4);
  (void)ws_size; (void)in_sizes; (void)n_in; (void)out_size;

  auto cdiv = [](int a, int b) { return (a + b - 1) / b; };
  for (int l = 0; l < 2; ++l) {
    pack_bf16<<<cdiv(272 * 288, 256), 256, 0, stream>>>(Wq + (size_t)l * 272 * 272, wq_bf + (size_t)l * 272 * 288, 272, 272, 272, 288);
    pack_bf16_T<<<cdiv(448 * 288, 256), 256, 0, stream>>>(Wk + (size_t)l * 272 * 444, wkT_bf + (size_t)l * 448 * 288, 272, 444, 448, 288);
    pack_bf16<<<cdiv(272 * 448, 256), 256, 0, stream>>>(Wv + (size_t)l * 272 * 444, wv_bf + (size_t)l * 272 * 448, 272, 272, 444, 448);
    pack_bf16<<<cdiv(272 * 288, 256), 256, 0, stream>>>(Wo + (size_t)l * 272 * 272, wo_bf + (size_t)l * 272 * 288, 272, 272, 272, 288);
    pack_bf16<<<cdiv(176 * 448, 256), 256, 0, stream>>>(W1 + (size_t)l * 172 * 444, w1_bf + (size_t)l * 176 * 448, 172, 176, 444, 448);
    pack_bf16<<<cdiv(176 * 192, 256), 256, 0, stream>>>(W2 + (size_t)l * 172 * 172, w2_bf + (size_t)l * 176 * 192, 172, 176, 172, 192);
  }

  hipFuncSetAttribute((const void*)tgn_agg,
                      hipFuncAttributeMaxDynamicSharedMemorySize, SM_TOTAL);

  // layer 1: 20000 queries (1-hop neighbors aggregating 2-hop), layer-0 weights
  tgn_agg<<<20000 / QB, NTHREADS, SM_TOTAL, stream>>>(
      20000, NN, 1, memory, edgef, memory, nbr1, tstmp,
      nbr2, eidx2, etim2, w_time, b_time,
      wq_bf, bq, wkT_bf, bk, wv_bf, bv, wo_bf, bo, w1_bf, b1, w2_bf, b2, emb1);

  // layer 2: 1000 source queries over emb1, layer-1 weights
  tgn_agg<<<1000 / QB, NTHREADS, SM_TOTAL, stream>>>(
      1000, 1, 0, memory, edgef, emb1, srcN, tstmp,
      nbr1, eidx1, etim1, w_time, b_time,
      wq_bf + (size_t)272 * 288, bq + 272, wkT_bf + (size_t)448 * 288, bk + 272,
      wv_bf + (size_t)272 * 448, bv + 272, wo_bf + (size_t)272 * 288, bo + 272,
      w1_bf + (size_t)176 * 448, b1 + 172, w2_bf + (size_t)176 * 192, b2 + 172, out);
}
